// LSTMCellWithProjection_53094385713708
// MI455X (gfx1250) — compile-verified, matrix operand order checked by reading, not run on hardware
//
#include <hip/hip_runtime.h>
#include <hip/hip_bf16.h>

// Problem constants (match reference)
#define BB 32
#define TT 512
#define DD 512
#define HH 512
#define CC 4096
#define FOURC (4 * CC)
#define K1 (DD + HH)   // 1024: fused concat(x_t, h) reduction dim

typedef __bf16 bf16_t;
typedef __attribute__((ext_vector_type(16))) __bf16 v16bf;
typedef __attribute__((ext_vector_type(8)))  float  v8f;

__device__ __forceinline__ float sigmoidf_(float x) { return 1.0f / (1.0f + __expf(-x)); }
__device__ __forceinline__ float clip3_(float x)    { return fminf(fmaxf(x, -3.0f), 3.0f); }

#define WMMA_BF16(A, B, C) \
  __builtin_amdgcn_wmma_f32_16x16x32_bf16(false, (A), false, (B), (short)0, (C), false, false)

// ---------------- setup kernels (run once per launch) ----------------

// W_cat[g, 0:512] = W_in[g, :], W_cat[g, 512:1024] = W_state[g, :]   (bf16)
__global__ void k_convert_wcat(const float* __restrict__ W_in,
                               const float* __restrict__ W_state,
                               bf16_t* __restrict__ Wcat) {
  const int total = FOURC * K1;
  for (int idx = blockIdx.x * blockDim.x + threadIdx.x; idx < total;
       idx += gridDim.x * blockDim.x) {
    int g = idx >> 10;          // row (gate-cell index)
    int k = idx & (K1 - 1);     // col
    float v = (k < DD) ? W_in[(size_t)g * DD + k]
                       : W_state[(size_t)g * HH + (k - DD)];
    Wcat[idx] = (bf16_t)v;
  }
}

__global__ void k_convert_wproj(const float* __restrict__ W_proj,
                                bf16_t* __restrict__ Wp) {
  const int total = HH * CC;
  for (int idx = blockIdx.x * blockDim.x + threadIdx.x; idx < total;
       idx += gridDim.x * blockDim.x) {
    Wp[idx] = (bf16_t)W_proj[idx];
  }
}

// Pre-convert ALL x timesteps to bf16: Zx[t][b][d] = bf16(x[b][t][d]).
__global__ void k_pack_x(const float* __restrict__ x, bf16_t* __restrict__ Zx) {
  const size_t total = (size_t)TT * BB * DD;
  for (size_t idx = (size_t)blockIdx.x * blockDim.x + threadIdx.x; idx < total;
       idx += (size_t)gridDim.x * blockDim.x) {
    size_t k = idx % DD;
    size_t r = idx / DD;
    size_t m = r % BB;
    size_t t = r / BB;
    Zx[idx] = (bf16_t)x[(m * TT + t) * DD + k];
  }
}

__global__ void k_zero_states(float* __restrict__ h_state,
                              float* __restrict__ c_state,
                              bf16_t* __restrict__ Zh) {
  int idx = blockIdx.x * blockDim.x + threadIdx.x;
  if (idx < BB * HH) { h_state[idx] = 0.0f; Zh[idx] = (bf16_t)0.0f; }
  if (idx < BB * CC) c_state[idx] = 0.0f;
}

// ---------------- per-timestep kernels ----------------

// Fused gate GEMM + nonlinearities + cell update.
// grid.x = CC/16 (cell tile), block = 128 (4 waves, one per gate).
// Each wave computes BOTH m-tiles (2 accumulators) sharing one B fragment,
// halving W_cat traffic: 2 WMMAs per weight-fragment load.
__global__ __launch_bounds__(128) void k_gates(
    const bf16_t* __restrict__ Zx,     // [TT, BB, DD] bf16 inputs
    const bf16_t* __restrict__ Zh,     // [BB, HH] bf16 hidden
    const bf16_t* __restrict__ Wcat,   // [FOURC, K1] row-major
    const float*  __restrict__ bias,   // [FOURC]
    float*        __restrict__ c_state,// [BB, CC]
    bf16_t*       __restrict__ y,      // [BB, CC] (o * tanh(mem))
    const long long* __restrict__ lens, int t) {
  __shared__ float lds[4][BB][16];

  const int ct   = blockIdx.x;            // cell tile: columns ct*16 .. +15
  const int gate = threadIdx.x >> 5;      // 0..3
  const int lane = threadIdx.x & 31;
  const int n0   = gate * CC + ct * 16;   // column into the 4C gate axis
  const int lrow = lane & 15;
  const int lkh  = lane >> 4;             // K-half select for fragments

  // Seed both accumulators with bias (depends only on column = lane&15).
  float bval = bias[n0 + lrow];
  v8f acc0, acc1;
#pragma unroll
  for (int v = 0; v < 8; ++v) { acc0[v] = bval; acc1[v] = bval; }

  const bf16_t* bptr = Wcat + (size_t)(n0 + lrow) * K1 + lkh * 16;

  // K segment 1: x contribution (K = 0..DD). 512 % 32 == 0, no straddle.
  {
    const bf16_t* a0 = Zx + ((size_t)t * BB + lrow) * DD + lkh * 16;
    const bf16_t* a1 = Zx + ((size_t)t * BB + 16 + lrow) * DD + lkh * 16;
    for (int k0 = 0; k0 < DD; k0 += 32) {
      v16bf bv  = *(const v16bf*)(bptr + k0);
      v16bf av0 = *(const v16bf*)(a0 + k0);
      v16bf av1 = *(const v16bf*)(a1 + k0);
      acc0 = WMMA_BF16(av0, bv, acc0);
      acc1 = WMMA_BF16(av1, bv, acc1);
    }
  }
  // K segment 2: h contribution (K = DD..K1).
  {
    const bf16_t* a0 = Zh + (size_t)lrow * HH + lkh * 16;
    const bf16_t* a1 = Zh + (size_t)(16 + lrow) * HH + lkh * 16;
    for (int k0 = 0; k0 < HH; k0 += 32) {
      v16bf bv  = *(const v16bf*)(bptr + DD + k0);
      v16bf av0 = *(const v16bf*)(a0 + k0);
      v16bf av1 = *(const v16bf*)(a1 + k0);
      acc0 = WMMA_BF16(av0, bv, acc0);
      acc1 = WMMA_BF16(av1, bv, acc1);
    }
  }

  // Stage f32 tiles to LDS so all 4 gates meet elementwise.
#pragma unroll
  for (int v = 0; v < 8; ++v) {
    int row = (lane < 16) ? v : (v + 8);
    lds[gate][row][lrow]      = acc0[v];
    lds[gate][16 + row][lrow] = acc1[v];
  }
  __syncthreads();

  // Elementwise epilogue: 32x16 cells, 128 threads -> 4 each.
  for (int e = threadIdx.x; e < BB * 16; e += 128) {
    int m  = e >> 4;                 // batch
    int cc = e & 15;
    int ci = ct * 16 + cc;           // cell index
    float ig = sigmoidf_(lds[0][m][cc]);
    float fg = sigmoidf_(lds[1][m][cc]);
    float gg = tanhf(lds[2][m][cc]);
    float og = sigmoidf_(lds[3][m][cc]);
    float cprev = c_state[m * CC + ci];
    float mem   = clip3_(ig * gg + fg * cprev);
    bool  msk   = (long long)t < lens[m];
    c_state[m * CC + ci] = msk ? mem : cprev;
    y[m * CC + ci] = (bf16_t)(og * tanhf(mem));
  }
}

// Projection GEMM + clip + masked h update + output + bf16 h for next step.
// grid.x = HH/16 = 32 (one n-tile per WG), block = 128 (4 waves).
// Each wave: one K-quarter (1024), both m-tiles (shared B fragment).
// Partial sums reduced through LDS.
__global__ __launch_bounds__(128) void k_proj(
    const bf16_t* __restrict__ y,       // [BB, CC]
    const bf16_t* __restrict__ Wp,      // [HH, CC] row-major
    float*        __restrict__ h_state, // [BB, HH]
    bf16_t*       __restrict__ Zh,      // [BB, HH] bf16 (next step's A)
    float*        __restrict__ out,     // [BB, TT, HH]
    const long long* __restrict__ lens, int t) {
  __shared__ float lds[4][2][16][16];   // [kchunk][mtile][row][col]

  const int nt    = blockIdx.x;          // 0..31
  const int w     = threadIdx.x >> 5;    // K chunk 0..3
  const int lane  = threadIdx.x & 31;
  const int n0    = nt * 16;
  const int lrow  = lane & 15;
  const int lkh   = lane >> 4;
  const int kbase = w * (CC / 4);        // 1024 per chunk

  v8f acc0 = {}, acc1 = {};
  const bf16_t* bptr = Wp + (size_t)(n0 + lrow) * CC + kbase + lkh * 16;
  const bf16_t* a0p  = y + (size_t)lrow * CC + kbase + lkh * 16;
  const bf16_t* a1p  = y + (size_t)(16 + lrow) * CC + kbase + lkh * 16;

  for (int k0 = 0; k0 < CC / 4; k0 += 32) {
    v16bf bv  = *(const v16bf*)(bptr + k0);
    v16bf av0 = *(const v16bf*)(a0p + k0);
    v16bf av1 = *(const v16bf*)(a1p + k0);
    acc0 = WMMA_BF16(av0, bv, acc0);
    acc1 = WMMA_BF16(av1, bv, acc1);
  }

#pragma unroll
  for (int v = 0; v < 8; ++v) {
    int row = (lane < 16) ? v : (v + 8);
    lds[w][0][row][lrow] = acc0[v];
    lds[w][1][row][lrow] = acc1[v];
  }
  __syncthreads();

  // Reduce 4 K-chunks + epilogue: 32x16 elements, 128 threads -> 4 each.
  for (int e = threadIdx.x; e < BB * 16; e += 128) {
    int m   = e >> 4;                 // batch 0..31
    int col = e & 15;
    int mt  = m >> 4, mr = m & 15;
    float s = lds[0][mt][mr][col] + lds[1][mt][mr][col] +
              lds[2][mt][mr][col] + lds[3][mt][mr][col];
    int n = n0 + col;
    float hh = clip3_(s);
    float hp = h_state[m * HH + n];
    bool  msk = (long long)t < lens[m];
    float hn = msk ? hh : hp;
    h_state[m * HH + n] = hn;
    Zh[m * HH + n] = (bf16_t)hn;
    out[((size_t)m * TT + t) * HH + n] = hn;
  }
}

// Copy final (h, c) into the tuple tail of d_out.
__global__ void k_finalize(const float* __restrict__ h_state,
                           const float* __restrict__ c_state,
                           float* __restrict__ out) {
  int idx = blockIdx.x * blockDim.x + threadIdx.x;
  const size_t base = (size_t)BB * TT * HH;
  if (idx < BB * HH) out[base + idx] = h_state[idx];
  if (idx < BB * CC) out[base + BB * HH + idx] = c_state[idx];
}

// ---------------- launcher ----------------

extern "C" void kernel_launch(void* const* d_in, const int* in_sizes, int n_in,
                              void* d_out, int out_size, void* d_ws, size_t ws_size,
                              hipStream_t stream) {
  const float*     x       = (const float*)d_in[0];      // [B, T, D] f32
  const long long* lens    = (const long long*)d_in[1];  // [B] int64
  const float*     W_in    = (const float*)d_in[2];      // [4C, D]
  const float*     W_state = (const float*)d_in[3];      // [4C, H]
  const float*     b_state = (const float*)d_in[4];      // [4C]
  const float*     W_proj  = (const float*)d_in[5];      // [H, C]
  float*           out     = (float*)d_out;

  char* ws = (char*)d_ws;
  bf16_t* Wcat    = (bf16_t*)ws; ws += (size_t)FOURC * K1 * sizeof(bf16_t);   // 33.5 MB
  bf16_t* Wp      = (bf16_t*)ws; ws += (size_t)HH * CC * sizeof(bf16_t);      //  4 MB
  bf16_t* Zx      = (bf16_t*)ws; ws += (size_t)TT * BB * DD * sizeof(bf16_t); // 16 MB
  float*  h_state = (float*)ws;  ws += (size_t)BB * HH * sizeof(float);
  float*  c_state = (float*)ws;  ws += (size_t)BB * CC * sizeof(float);
  bf16_t* Zh      = (bf16_t*)ws; ws += (size_t)BB * HH * sizeof(bf16_t);
  bf16_t* y       = (bf16_t*)ws;

  // One-time conversions (bf16 weights stay L2-resident across all 512 steps).
  k_convert_wcat<<<8192, 256, 0, stream>>>(W_in, W_state, Wcat);
  k_convert_wproj<<<2048, 256, 0, stream>>>(W_proj, Wp);
  k_pack_x<<<8192, 256, 0, stream>>>(x, Zx);
  k_zero_states<<<(BB * CC + 255) / 256, 256, 0, stream>>>(h_state, c_state, Zh);

  // Sequential recurrence: 2 kernels per timestep.
  for (int t = 0; t < TT; ++t) {
    k_gates<<<CC / 16, 128, 0, stream>>>(Zx, Zh, Wcat, b_state, c_state, y, lens, t);
    k_proj<<<HH / 16, 128, 0, stream>>>(y, Wp, h_state, Zh, out, lens, t);
  }

  k_finalize<<<(BB * CC + 255) / 256, 256, 0, stream>>>(h_state, c_state, out);
}
